// LFSAM_25357486916393
// MI455X (gfx1250) — matrix operand order repaired and checked by reading, not compiled, not run on hardware
//
#include <hip/hip_runtime.h>

// MI455X / gfx1250 flash-style fused self-attention (SAGAN block).
// B=4, H=W=64, C=64, C8=8, N=4096.
// Pipeline:
//   K1: 1x1-conv projections f,g (B,N,8 f16) and h (B,C,N f16), relu'd.
//   K2: per-row softmax stats (max, 1/sum) of s = g^T f via v_wmma_f32_16x16x32_f16,
//       computed on transposed tiles so the reduction is per-lane.
//   K3: o = h @ P with P = exp(s - max)*inv recomputed tile-by-tile (flash style),
//       P converted D->B layout through LDS, 4 accumulator WMMA tiles per wave,
//       fused epilogue y = gamma*o + x with the inverse (0,3,2,1) permute.

typedef __attribute__((ext_vector_type(16))) _Float16 v16h;
typedef __attribute__((ext_vector_type(8)))  float    v8f;

#define BQ 4
#define HQ 64
#define WQ 64
#define CQ 64
#define C8Q 8
#define NQ 4096

union HVec {
    v16h v;
    unsigned int u[8];
    _Float16 h[16];
};

// ---------------------------------------------------------------- K1: projections
__global__ void lfsam_proj(const float* __restrict__ x,
                           const float* __restrict__ Wf, const float* __restrict__ bf,
                           const float* __restrict__ Wg, const float* __restrict__ bg,
                           const float* __restrict__ Wh, const float* __restrict__ bh,
                           _Float16* __restrict__ f_t,   // [B][N][8]
                           _Float16* __restrict__ g_t,   // [B][N][8]
                           _Float16* __restrict__ h_h)   // [B][C][N]
{
    __shared__ float xs[CQ];
    const int p   = blockIdx.x;          // 0 .. B*N-1
    const int b   = p >> 12;
    const int n   = p & (NQ - 1);        // n = w*64 + h  (xp flatten order)
    const int hh  = n & 63;
    const int ww  = n >> 6;
    const int tid = threadIdx.x;         // 0..63

    const float* xrow = x + (((size_t)(b * HQ + hh) * WQ + ww) * CQ);
    xs[tid] = xrow[tid];
    __syncthreads();

    // h[d] = relu(sum_c Wh[c,d] * x[c] + bh[d]), d = tid
    float ah = bh[tid];
    #pragma unroll 8
    for (int c = 0; c < CQ; ++c) ah = fmaf(xs[c], Wh[c * CQ + tid], ah);
    ah = ah > 0.f ? ah : 0.f;
    h_h[((size_t)(b * CQ + tid)) * NQ + n] = (_Float16)ah;

    // f (tid 0..7) and g (tid 8..15)
    if (tid < 16) {
        const int k = tid & 7;
        const float* Wp = (tid < 8) ? Wf : Wg;
        float a = (tid < 8) ? bf[k] : bg[k];
        #pragma unroll 8
        for (int c = 0; c < CQ; ++c) a = fmaf(xs[c], Wp[c * C8Q + k], a);
        a = a > 0.f ? a : 0.f;
        _Float16* dst = (tid < 8) ? f_t : g_t;
        dst[((size_t)(b * NQ + n)) * C8Q + k] = (_Float16)a;
    }
}

// Load an 8-wide f16 row as a (K<=8) WMMA operand: lanes 0-15 carry K=0..7,
// everything else zero (valid as both A and B operand when K_actual == 8).
__device__ inline HVec load_k8_operand(const _Float16* __restrict__ base,
                                       int row, unsigned int msk)
{
    HVec r;
    const uint4 q = *reinterpret_cast<const uint4*>(base + (size_t)row * C8Q);
    r.u[0] = q.x & msk; r.u[1] = q.y & msk; r.u[2] = q.z & msk; r.u[3] = q.w & msk;
    r.u[4] = 0u; r.u[5] = 0u; r.u[6] = 0u; r.u[7] = 0u;
    return r;
}

// ---------------------------------------------------------------- K2: softmax stats
// One wave per 16 consecutive k rows of s[k,m]; scans all m tiles.
// Tile T[m,k] = sum_c f[c,m] g[c,k] == s[k,m]; per-lane online (max, sum),
// final shfl_xor(16) merges the two row-halves of each column.
__global__ void lfsam_stats(const _Float16* __restrict__ f_t,
                            const _Float16* __restrict__ g_t,
                            float* __restrict__ rowmax,
                            float* __restrict__ rowinv)
{
    const int wid  = blockIdx.x;         // 0 .. B*(N/16)-1
    const int b    = wid >> 8;
    const int k0   = (wid & 255) * 16;
    const int lane = threadIdx.x;        // 0..31
    const int Lm   = lane & 15;
    const unsigned int msk = (lane < 16) ? 0xFFFFFFFFu : 0u;

    const _Float16* fb = f_t + (size_t)b * NQ * C8Q;
    const _Float16* gb = g_t + (size_t)b * NQ * C8Q;

    const HVec gB = load_k8_operand(gb, k0 + Lm, msk);  // B: cols k0..k0+15

    float run_m = -__builtin_inff();
    float run_s = 0.f;

    for (int mt = 0; mt < NQ / 16; ++mt) {
        const HVec fA = load_k8_operand(fb, mt * 16 + Lm, msk);  // A: rows m
        v8f z = {};
        v8f d = __builtin_amdgcn_wmma_f32_16x16x32_f16(
            false, fA.v, false, gB.v, (short)0, z, false, false);

        float tmax = d[0];
        #pragma unroll
        for (int r = 1; r < 8; ++r) tmax = fmaxf(tmax, d[r]);
        const float nm = fmaxf(run_m, tmax);
        float s = run_s * __expf(run_m - nm);
        #pragma unroll
        for (int r = 0; r < 8; ++r) s += __expf(d[r] - nm);
        run_s = s;
        run_m = nm;
    }

    const float pm = __shfl_xor(run_m, 16, 32);
    const float ps = __shfl_xor(run_s, 16, 32);
    const float M  = fmaxf(run_m, pm);
    const float S  = run_s * __expf(run_m - M) + ps * __expf(pm - M);  // S >= 1
    if (lane < 16) {
        rowmax[b * NQ + k0 + lane] = M;
        rowinv[b * NQ + k0 + lane] = 1.0f / S;
    }
}

// ---------------------------------------------------------------- K3: output
// One wave per (b, 16-col n tile). Holds 4 accumulator tiles (all 64 c rows).
// Per 32-k step: recompute s tiles (2 WMMA), build P in LDS, 4 accumulate WMMAs.
__global__ void lfsam_out(const _Float16* __restrict__ f_t,
                          const _Float16* __restrict__ g_t,
                          const _Float16* __restrict__ h_h,
                          const float* __restrict__ rowmax,
                          const float* __restrict__ rowinv,
                          const float* __restrict__ x,
                          const float* __restrict__ gamma_p,
                          float* __restrict__ out)
{
    __shared__ _Float16 Pl[32][17];   // padded to dodge bank conflicts
    __shared__ float rmS[32];
    __shared__ float riS[32];

    const int wid  = blockIdx.x;         // 0 .. B*(N/16)-1
    const int b    = wid >> 8;
    const int n0   = (wid & 255) * 16;
    const int lane = threadIdx.x;        // 0..31
    const int Lm   = lane & 15;
    const bool lo  = (lane < 16);
    const unsigned int msk = lo ? 0xFFFFFFFFu : 0u;

    const _Float16* fb = f_t + (size_t)b * NQ * C8Q;
    const _Float16* gb = g_t + (size_t)b * NQ * C8Q;
    const _Float16* hb = h_h + (size_t)b * CQ * NQ;
    const float* rm = rowmax + b * NQ;
    const float* ri = rowinv + b * NQ;

    const HVec fB = load_k8_operand(fb, n0 + Lm, msk);  // loop-invariant B (cols n)

    v8f acc0 = {}, acc1 = {}, acc2 = {}, acc3 = {};

    const int rbase = lo ? 0 : 8;        // D-layout row base for this half-wave
    const int aoff  = lo ? 0 : 8;        // A-layout K offset for this half-wave
    const int cbB   = lo ? 0 : 8;        // B-layout K base when reading P

    for (int kt = 0; kt < NQ / 32; ++kt) {
        const int k0 = kt * 32;

        // --- score tiles s[k0..k0+31, n0..n0+15]
        const HVec gA1 = load_k8_operand(gb, k0 + Lm, msk);
        const HVec gA2 = load_k8_operand(gb, k0 + 16 + Lm, msk);
        v8f z = {};
        v8f d1 = __builtin_amdgcn_wmma_f32_16x16x32_f16(
            false, gA1.v, false, fB.v, (short)0, z, false, false);
        v8f d2 = __builtin_amdgcn_wmma_f32_16x16x32_f16(
            false, gA2.v, false, fB.v, (short)0, z, false, false);

        // --- stage softmax stats for these 32 k rows
        rmS[lane] = rm[k0 + lane];
        riS[lane] = ri[k0 + lane];
        __syncthreads();

        // --- P = exp(s - max) * inv, D-layout -> LDS (32 x 16, f16)
        #pragma unroll
        for (int r = 0; r < 8; ++r) {
            const int kr  = rbase + r;
            const int kr2 = 16 + kr;
            Pl[kr][Lm]  = (_Float16)(__expf(d1[r] - rmS[kr])  * riS[kr]);
            Pl[kr2][Lm] = (_Float16)(__expf(d2[r] - rmS[kr2]) * riS[kr2]);
        }
        __syncthreads();

        // --- read P back in B-operand layout (32x16 f16)
        HVec pB;
        #pragma unroll
        for (int j = 0; j < 8; ++j) {
            pB.h[j]     = Pl[cbB + j][Lm];
            pB.h[j + 8] = Pl[cbB + 16 + j][Lm];
        }

        // --- 4 accumulate tiles: A = h[c_tile, k0..k0+31]
        #pragma unroll
        for (int t = 0; t < 4; ++t) {
            const _Float16* hr = hb + (size_t)(16 * t + Lm) * NQ + k0;
            HVec hA;
            const uint4 qa = *reinterpret_cast<const uint4*>(hr + aoff);       // K 0-7 / 8-15
            const uint4 qb = *reinterpret_cast<const uint4*>(hr + aoff + 16);  // K 16-23 / 24-31
            hA.u[0] = qa.x; hA.u[1] = qa.y; hA.u[2] = qa.z; hA.u[3] = qa.w;
            hA.u[4] = qb.x; hA.u[5] = qb.y; hA.u[6] = qb.z; hA.u[7] = qb.w;
            v8f* ap = (t == 0) ? &acc0 : (t == 1) ? &acc1 : (t == 2) ? &acc2 : &acc3;
            *ap = __builtin_amdgcn_wmma_f32_16x16x32_f16(
                false, hA.v, false, pB.v, (short)0, *ap, false, false);
        }
    }

    // --- epilogue: y[b,h,w,c] = gamma * o[b,c,n] + x[b,h,w,c],  n = w*64 + h
    const float gm = gamma_p[0];
    const int nn = n0 + Lm;
    const int hh = nn & 63;
    const int ww = nn >> 6;
    const size_t base = ((size_t)(b * HQ + hh) * WQ + ww) * CQ;
    const int hi8 = lo ? 0 : 8;

    auto emit = [&](const v8f& a, int t) {
        const size_t i0 = base + (size_t)(16 * t + hi8);
        #pragma unroll
        for (int r = 0; r < 8; ++r)
            out[i0 + r] = fmaf(gm, a[r], x[i0 + r]);
    };
    emit(acc0, 0); emit(acc1, 1); emit(acc2, 2); emit(acc3, 3);
}

// ---------------------------------------------------------------- launcher
extern "C" void kernel_launch(void* const* d_in, const int* in_sizes, int n_in,
                              void* d_out, int out_size, void* d_ws, size_t ws_size,
                              hipStream_t stream) {
    (void)in_sizes; (void)n_in; (void)out_size; (void)ws_size;
    const float* x     = (const float*)d_in[0];
    const float* Wf    = (const float*)d_in[1];
    const float* bf    = (const float*)d_in[2];
    const float* Wg    = (const float*)d_in[3];
    const float* bg    = (const float*)d_in[4];
    const float* Wh    = (const float*)d_in[5];
    const float* bh    = (const float*)d_in[6];
    const float* gamma = (const float*)d_in[7];
    float* out = (float*)d_out;

    char* ws = (char*)d_ws;
    _Float16* f_t  = (_Float16*)(ws);                     // 4*4096*8*2   = 256 KB
    _Float16* g_t  = (_Float16*)(ws + (256u << 10));      // 256 KB
    _Float16* h_h  = (_Float16*)(ws + (512u << 10));      // 4*64*4096*2  = 2 MB
    float*    rmax = (float*)   (ws + (2560u << 10));     // 64 KB
    float*    rinv = (float*)   (ws + (2624u << 10));     // 64 KB  (total ~2.63 MB)

    lfsam_proj<<<BQ * NQ, CQ, 0, stream>>>(x, Wf, bf, Wg, bg, Wh, bh, f_t, g_t, h_h);
    lfsam_stats<<<BQ * (NQ / 16), 32, 0, stream>>>(f_t, g_t, rmax, rinv);
    lfsam_out<<<BQ * (NQ / 16), 32, 0, stream>>>(f_t, g_t, h_h, rmax, rinv, x, gamma, out);
}
